// MLPNN_41351945126312
// MI455X (gfx1250) — matrix operand order repaired and checked by reading, not compiled, not run on hardware
//
#include <hip/hip_runtime.h>
#include <math.h>

#define N_NODE 16384
#define HID    64

typedef __attribute__((ext_vector_type(16))) _Float16 v16h;
typedef __attribute__((ext_vector_type(8)))  _Float16 v8h;
typedef __attribute__((ext_vector_type(8)))  float    v8f;

// ---------------------------------------------------------------------------
// 1) LayerNorm: one wave32 per row (lane handles d and d+32)
// ---------------------------------------------------------------------------
__global__ __launch_bounds__(256) void hgnn_ln_kernel(
    const float* __restrict__ vemb, const float* __restrict__ gamma,
    const float* __restrict__ beta, float* __restrict__ h)
{
  const int wave = threadIdx.x >> 5;
  const int lane = threadIdx.x & 31;
  const int row  = blockIdx.x * 8 + wave;
  const float* x = vemb + (size_t)row * HID;
  float a0 = x[lane];
  float a1 = x[lane + 32];
  float s = a0 + a1;
  float q = a0 * a0 + a1 * a1;
  #pragma unroll
  for (int off = 16; off > 0; off >>= 1) {
    s += __shfl_xor(s, off, 32);
    q += __shfl_xor(q, off, 32);
  }
  const float mu   = s * (1.0f / HID);
  const float var  = q * (1.0f / HID) - mu * mu;
  const float rstd = rsqrtf(var + 1e-5f);
  float* hr = h + (size_t)row * HID;
  hr[lane]      = (a0 - mu) * rstd * gamma[lane]      + beta[lane];
  hr[lane + 32] = (a1 - mu) * rstd * gamma[lane + 32] + beta[lane + 32];
}

// ---------------------------------------------------------------------------
// 2) CSR row_ptr from sorted adj_row (np.unique output is sorted)
// ---------------------------------------------------------------------------
__global__ void hgnn_rowptr_kernel(const int* __restrict__ adj_row, int nnz,
                                   int* __restrict__ row_ptr)
{
  const int r = blockIdx.x * blockDim.x + threadIdx.x;
  if (r > N_NODE) return;
  int lo = 0, hi = nnz;
  while (lo < hi) {
    int mid = (lo + hi) >> 1;
    if (adj_row[mid] < r) lo = mid + 1; else hi = mid;
  }
  row_ptr[r] = lo;
}

// ---------------------------------------------------------------------------
// 3) Broadcast h into all T copies of xv
// ---------------------------------------------------------------------------
__global__ __launch_bounds__(256) void hgnn_init_xv_kernel(
    const float* __restrict__ h, float* __restrict__ xv)
{
  const size_t i = (size_t)blockIdx.x * 256 + threadIdx.x;  // over N*HID
  xv[(size_t)blockIdx.y * ((size_t)N_NODE * HID) + i] = h[i];
}

// ---------------------------------------------------------------------------
// 4) Per-copy labeled-row fixup: xv[t, tar[t], :] = h[tar[t]] @ W_lab + b_lab
// ---------------------------------------------------------------------------
__global__ void hgnn_lab_kernel(const float* __restrict__ h,
                                const float* __restrict__ W_lab,
                                const float* __restrict__ b_lab,
                                const int* __restrict__ tar,
                                float* __restrict__ xv)
{
  const int t = blockIdx.x;
  const int d = threadIdx.x;
  const int node = tar[t];
  const float* hr = h + (size_t)node * HID;
  float acc = b_lab[d];
  #pragma unroll 8
  for (int k = 0; k < HID; ++k) acc += hr[k] * W_lab[k * HID + d];
  xv[((size_t)t * N_NODE + node) * HID + d] = acc;
}

// ---------------------------------------------------------------------------
// 5) WMMA GEMM: y16[t] = f16( xv[t] @ W_lin + b_lin )
//    block = 256 threads (8 waves); block tile = 64 rows x 64 cols.
//    Wave w owns row-tile (w>>1) and TWO adjacent 16-col tiles at (w&1)*32:
//    two independent accumulators interleave v_wmma issue (no RAW v_nops)
//    and share the A fragment. LDS layouts are chosen so every fragment is
//    contiguous ds_load_b128 chunks (A and B share one k order):
//      A 16x32: a[0..15]=K kb+16*hi..+15 for fixed M  (row-major As)
//      B 32x16: b[0..15]=K kb+16*hi..+15 for fixed N          (store W^T)
// ---------------------------------------------------------------------------
__global__ __launch_bounds__(256) void hgnn_gemm_wmma_kernel(
    const float* __restrict__ xv, const float* __restrict__ W,
    const float* __restrict__ bias, _Float16* __restrict__ y)
{
  __shared__ __attribute__((aligned(16))) _Float16 As[64][HID + 8]; // x tile
  __shared__ __attribute__((aligned(16))) _Float16 Bt[64][HID + 8]; // W^T

  const int    t    = blockIdx.y;
  const int    row0 = blockIdx.x * 64;
  const size_t base = ((size_t)t * N_NODE + row0) * HID;

  // Stage X tile (64x64) as f16, 8 contiguous halves per chunk -> b128 stores
  for (int i = threadIdx.x; i < 64 * 8; i += 256) {
    const int m = i >> 3, cj = i & 7;
    const float* src = &xv[base + (size_t)m * HID + cj * 8];
    v8h tmp;
    #pragma unroll
    for (int j = 0; j < 8; ++j) tmp[j] = (_Float16)src[j];
    ((v8h*)&As[m][0])[cj] = tmp;
  }
  // Stage W transposed (Bt[n][k] = W[k][n]) as f16
  for (int i = threadIdx.x; i < HID * HID; i += 256) {
    const int k = i >> 6, n = i & 63;
    Bt[n][k] = (_Float16)W[i];
  }
  __syncthreads();

  const int lane = threadIdx.x & 31;
  const int wave = threadIdx.x >> 5;
  const int m0   = (wave >> 1) << 4;        // 0,16,32,48
  const int n0   = (wave & 1) << 5;         // 0 or 32: two 16-col tiles
  const int l15  = lane & 15;
  const int hi   = lane >> 4;

  v8f c0, c1;
  const float bv0 = bias[n0 + l15];         // C layout: lane&15 == N
  const float bv1 = bias[n0 + 16 + l15];
  #pragma unroll
  for (int r = 0; r < 8; ++r) { c0[r] = bv0; c1[r] = bv1; }

  const v8h* Arow  = (const v8h*)&As[m0 + l15][0];
  const v8h* Brow0 = (const v8h*)&Bt[n0 + l15][0];
  const v8h* Brow1 = (const v8h*)&Bt[n0 + 16 + l15][0];

  #pragma unroll
  for (int kb = 0; kb < HID; kb += 32) {
    const int c8 = kb >> 3;                 // v8h chunk index of this k-block
    union { v16h v; v8h h[2]; } ua, ub0, ub1;
    ua.h[0]  = Arow[c8 + 2 * hi];           // K = kb+16*hi   .. +7
    ua.h[1]  = Arow[c8 + 2 * hi + 1];       // K = kb+16*hi+8 .. +15
    ub0.h[0] = Brow0[c8 + 2 * hi];          // K = kb+16*hi   .. +7
    ub0.h[1] = Brow0[c8 + 2 * hi + 1];      // K = kb+16*hi+8 .. +15
    ub1.h[0] = Brow1[c8 + 2 * hi];
    ub1.h[1] = Brow1[c8 + 2 * hi + 1];
    c0 = __builtin_amdgcn_wmma_f32_16x16x32_f16(false, ua.v, false, ub0.v,
                                                (short)0, c0, false, false);
    c1 = __builtin_amdgcn_wmma_f32_16x16x32_f16(false, ua.v, false, ub1.v,
                                                (short)0, c1, false, false);
  }

  // Store f16 results per C layout: M = r + 8*hi, N = lane&15
  #pragma unroll
  for (int r = 0; r < 8; ++r) {
    const int mm = m0 + r + (hi << 3);
    _Float16* dst = &y[base + (size_t)mm * HID + l15];
    dst[n0]      = (_Float16)c0[r];
    dst[n0 + 16] = (_Float16)c1[r];
  }
}

// ---------------------------------------------------------------------------
// 6) Row segment-max over scaled gathered columns + residual add
//    block = 256 threads = 4 rows x 64 dims
// ---------------------------------------------------------------------------
__global__ __launch_bounds__(256) void hgnn_spmm_max_kernel(
    const _Float16* __restrict__ y, const float* __restrict__ adj_val,
    const int* __restrict__ adj_col, const int* __restrict__ row_ptr,
    float* __restrict__ xv)
{
  const int d = threadIdx.x & 63;
  const int r = blockIdx.x * 4 + (threadIdx.x >> 6);
  const int t = blockIdx.y;
  const int s = row_ptr[r];
  const int e = row_ptr[r + 1];
  const _Float16* yb = y + (size_t)t * N_NODE * HID;
  float m = -INFINITY;
  for (int i = s; i < e; ++i) {
    if (i + 1 < e)
      __builtin_prefetch(&yb[(size_t)adj_col[i + 1] * HID + d], 0, 0);
    m = fmaxf(m, adj_val[i] * (float)yb[(size_t)adj_col[i] * HID + d]);
  }
  if (!isfinite(m)) m = 0.0f;  // empty rows -> 0 (matches jnp.where(isfinite))
  xv[((size_t)t * N_NODE + r) * HID + d] += m;
}

// ---------------------------------------------------------------------------
// 7) Final pair-max over groups of (K_TAR*K_TAR) pairs + dot with W_out
// ---------------------------------------------------------------------------
__global__ void hgnn_out_kernel(const float* __restrict__ xv,
                                const int* __restrict__ tar,
                                const int* __restrict__ pair_copy,
                                const int* __restrict__ pair_node,
                                const float* __restrict__ W_out,
                                const float* __restrict__ b_out,
                                float* __restrict__ out, int per_batch)
{
  __shared__ float red[HID];
  const int b = blockIdx.x;
  const int d = threadIdx.x;
  float m = -INFINITY;
  for (int j = 0; j < per_batch; ++j) {
    const int p    = b * per_batch + j;
    const int t    = pair_copy[p];
    const int node = tar[pair_node[p]];
    m = fmaxf(m, xv[((size_t)t * N_NODE + node) * HID + d]);
  }
  if (!isfinite(m)) m = 0.0f;
  red[d] = m * W_out[d];
  __syncthreads();
  #pragma unroll
  for (int off = 32; off > 0; off >>= 1) {
    if (d < off) red[d] += red[d + off];
    __syncthreads();
  }
  if (d == 0) out[b] = red[0] + b_out[0];
}

// ---------------------------------------------------------------------------
extern "C" void kernel_launch(void* const* d_in, const int* in_sizes, int n_in,
                              void* d_out, int out_size, void* d_ws, size_t ws_size,
                              hipStream_t stream)
{
  (void)n_in; (void)ws_size;
  const float* vemb      = (const float*)d_in[0];
  const float* ln_gamma  = (const float*)d_in[1];
  const float* ln_beta   = (const float*)d_in[2];
  const float* W_lab     = (const float*)d_in[3];
  const float* b_lab     = (const float*)d_in[4];
  const float* W_lin     = (const float*)d_in[5];
  const float* b_lin     = (const float*)d_in[6];
  const float* W_out     = (const float*)d_in[7];
  const float* b_out     = (const float*)d_in[8];
  const float* adj_val   = (const float*)d_in[9];
  const int*   adj_row   = (const int*)d_in[10];
  const int*   adj_col   = (const int*)d_in[11];
  const int*   tar_nodes = (const int*)d_in[12];
  const int*   pair_copy = (const int*)d_in[13];
  const int*   pair_node = (const int*)d_in[14];
  float*       out       = (float*)d_out;

  const int T       = in_sizes[12];              // unique target nodes (<= 32)
  const int nnz     = in_sizes[9];
  const int npairs  = in_sizes[13];
  const int num_tar = out_size;                  // 8
  const int per_b   = npairs / num_tar;          // 16

  // Workspace layout
  char* ws = (char*)d_ws;
  float*    h       = (float*)ws;                                   // 4 MB
  int*      row_ptr = (int*)(ws + (size_t)N_NODE * HID * 4);        // 64 KB+
  float*    xv      = (float*)(ws + (size_t)N_NODE * HID * 4 + (128u << 10));
  _Float16* y16     = (_Float16*)((char*)xv + (size_t)T * N_NODE * HID * 4);

  // 1) LayerNorm
  hgnn_ln_kernel<<<N_NODE / 8, 256, 0, stream>>>(vemb, ln_gamma, ln_beta, h);
  // 2) CSR row_ptr
  hgnn_rowptr_kernel<<<(N_NODE + 1 + 255) / 256, 256, 0, stream>>>(adj_row, nnz, row_ptr);
  // 3) broadcast init
  hgnn_init_xv_kernel<<<dim3(N_NODE * HID / 256, T), 256, 0, stream>>>(h, xv);
  // 4) labeled-row fixup
  hgnn_lab_kernel<<<T, HID, 0, stream>>>(h, W_lab, b_lab, tar_nodes, xv);
  // 5/6) two propagation layers
  for (int layer = 0; layer < 2; ++layer) {
    hgnn_gemm_wmma_kernel<<<dim3(N_NODE / 64, T), 256, 0, stream>>>(xv, W_lin, b_lin, y16);
    hgnn_spmm_max_kernel<<<dim3(N_NODE / 4, T), 256, 0, stream>>>(y16, adj_val, adj_col, row_ptr, xv);
  }
  // 7) readout
  hgnn_out_kernel<<<num_tar, HID, 0, stream>>>(xv, tar_nodes, pair_copy, pair_node,
                                               W_out, b_out, out, per_b);
}
